// TransformerEncoderLayer_Pola_SEFN_75453985457242
// MI455X (gfx1250) — matrix operand order repaired and checked by reading, not compiled, not run on hardware
//
#include <hip/hip_runtime.h>
#include <hip/hip_bf16.h>

// ---------------------------------------------------------------------------
// Problem constants (from reference): B=64, C=256, H=W=20, N=400, heads=8
// ---------------------------------------------------------------------------
#define BB   64
#define CC   256
#define HH   20
#define WW2  20
#define NN   400            // H*W
#define HEADS 8
#define HD   32             // C / heads
#define BN   (BB*NN)        // 25600
#define HID  512            // int(C*2.0)

typedef __attribute__((ext_vector_type(16))) _Float16 v16h;
typedef __attribute__((ext_vector_type(8)))  _Float16 v8h;
typedef __attribute__((ext_vector_type(8)))  float    v8f;

// ---------------------------------------------------------------------------
// Workspace layout (bytes). All sizes multiples of 256.
// ---------------------------------------------------------------------------
static constexpr size_t SZ_XH     = (size_t)BN*CC*2;          // x (BN,C) f16
static constexpr size_t SZ_WQGKV  = (size_t)1024*CC*2;        // [qg_w;kv_w] f16
static constexpr size_t SZ_WPROJ  = (size_t)CC*CC*2;
static constexpr size_t SZ_WPIN   = (size_t)1024*CC*2;
static constexpr size_t SZ_WPOUT  = (size_t)CC*HID*2;
static constexpr size_t SZ_WSE    = (size_t)HID*CC*2;
static constexpr size_t SZ_BIAS   = (size_t)1024*4;
static constexpr size_t SZ_ZB     = (size_t)1024*4;           // zero bias
static constexpr size_t SZ_GEMMB  = (size_t)BN*1024*4;        // qgkv out, later pin out
static constexpr size_t SZ_QP     = (size_t)BB*HEADS*NN*HD*4; // 26.2MB each
static constexpr size_t SZ_G      = (size_t)BN*CC*4;
static constexpr size_t SZ_KMEAN  = (size_t)BB*HEADS*64*4;
static constexpr size_t SZ_KVMAT  = (size_t)BB*HEADS*2048*4;
static constexpr size_t SZ_AH     = (size_t)BN*CC*2;
static constexpr size_t SZ_SMH    = (size_t)BB*CC*2;
static constexpr size_t SZ_SEPRE  = (size_t)BB*HID*4;
static constexpr size_t SZ_YH     = (size_t)BN*HID*2;

static constexpr size_t OFF_XH    = 0;
static constexpr size_t OFF_WQGKV = OFF_XH    + SZ_XH;
static constexpr size_t OFF_WPROJ = OFF_WQGKV + SZ_WQGKV;
static constexpr size_t OFF_WPIN  = OFF_WPROJ + SZ_WPROJ;
static constexpr size_t OFF_WPOUT = OFF_WPIN  + SZ_WPIN;
static constexpr size_t OFF_WSE   = OFF_WPOUT + SZ_WPOUT;
static constexpr size_t OFF_BIAS  = OFF_WSE   + SZ_WSE;
static constexpr size_t OFF_ZB    = OFF_BIAS  + SZ_BIAS;
static constexpr size_t OFF_GEMMB = OFF_ZB    + SZ_ZB;
static constexpr size_t OFF_QP    = OFF_GEMMB + SZ_GEMMB;
static constexpr size_t OFF_QN    = OFF_QP + SZ_QP;
static constexpr size_t OFF_KP    = OFF_QN + SZ_QP;
static constexpr size_t OFF_KN    = OFF_KP + SZ_QP;
static constexpr size_t OFF_G     = OFF_KN + SZ_QP;
static constexpr size_t OFF_VF    = OFF_G  + SZ_G;            // v (BN,C) f32
static constexpr size_t OFF_KMEAN = OFF_VF + SZ_G;
static constexpr size_t OFF_KVMAT = OFF_KMEAN + SZ_KMEAN;
static constexpr size_t OFF_AT    = OFF_KVMAT + SZ_KVMAT;     // attention out (BN,C)
static constexpr size_t OFF_VD    = OFF_AT + SZ_G;            // dwconv out (BN,C)
static constexpr size_t OFF_AH    = OFF_VD + SZ_G;            // (attn+vd)*g f16
static constexpr size_t OFF_PA    = OFF_AH + SZ_AH;           // proj out a, later f
static constexpr size_t OFF_S     = OFF_PA + SZ_G;            // s (B,C,N) f32
static constexpr size_t OFF_SH    = OFF_S  + SZ_G;            // s (BN,C) f16
static constexpr size_t OFF_SMH   = OFF_SH + SZ_AH;
static constexpr size_t OFF_SEPRE = OFF_SMH + SZ_SMH;
static constexpr size_t OFF_YH    = OFF_SEPRE + SZ_SEPRE;
// total ~439 MB

// ---------------------------------------------------------------------------
// WMMA GEMM:  C(M,Nn) = A(M,K) * W(Nn,K)^T + bias.
// Requirements: M%64==0, Nn%128==0, K%64==0, bias non-null (use zero buffer).
// Block: 256 threads = 8 waves; block tile 64x128 (M x N).
// Wave tile 32x32: 2 A-frags x 2 B-frags per 32-K slice, 2 slices per buffer
// -> 8 WMMAs per barrier.  Double-buffered LDS software pipeline: the next
// 64-K tile's global loads are issued before the WMMAs on the current tile;
// ds_store to the alternate buffer afterwards; one barrier per 64-K step.
// global_prefetch issued two tiles ahead.
// ---------------------------------------------------------------------------
__global__ __launch_bounds__(256) void k_wmma_gemm(
    const _Float16* __restrict__ A, const _Float16* __restrict__ W,
    const float* __restrict__ bias, float* __restrict__ Cout,
    int M, int Nn, int K)
{
  __shared__ _Float16 As[2][64][64];    // 2 x 8KB
  __shared__ _Float16 Ws[2][128][64];   // 2 x 16KB

  const int tid  = threadIdx.x;
  const int lane = tid & 31;
  const int wave = tid >> 5;
  const int hl   = lane & 15;
  const int hi   = lane >> 4;          // 0 | 1 half-wave
  const int wr   = wave >> 2;          // 0..1 : 32-row group
  const int wc   = wave & 3;           // 0..3 : 32-col group
  const int blockM = blockIdx.x * 64;
  const int blockN = blockIdx.y * 128;

  // global->LDS staging: 16B chunks; A tile 64x64 halves = 512 chunks (2/thread),
  // W tile 128x64 halves = 1024 chunks (4/thread).
  const int aRow0 = tid >> 3;          // 0..31   (chunk id = tid)
  const int aRow1 = (tid + 256) >> 3;  // 32..63  (chunk id = tid+256)
  const int aCol  = (tid & 7) * 8;     // 0..56
  const _Float16* Aptr0 = A + (size_t)(blockM + aRow0) * K + aCol;
  const _Float16* Aptr1 = A + (size_t)(blockM + aRow1) * K + aCol;
  const _Float16* Wp0 = W + (size_t)(blockN + aRow0) * K + aCol;
  const _Float16* Wp1 = W + (size_t)(blockN + aRow1) * K + aCol;
  const _Float16* Wp2 = W + (size_t)(blockN + aRow0 + 64) * K + aCol;
  const _Float16* Wp3 = W + (size_t)(blockN + aRow1 + 64) * K + aCol;

  v8f acc00 = {}, acc01 = {}, acc10 = {}, acc11 = {};

  const int nk = K >> 6;               // 64-wide K tiles

  // prologue: stage tile 0
  uint4 ra0 = *(const uint4*)Aptr0;
  uint4 ra1 = *(const uint4*)Aptr1;
  uint4 rw0 = *(const uint4*)Wp0;
  uint4 rw1 = *(const uint4*)Wp1;
  uint4 rw2 = *(const uint4*)Wp2;
  uint4 rw3 = *(const uint4*)Wp3;
  *(uint4*)&As[0][aRow0][aCol]      = ra0;
  *(uint4*)&As[0][aRow1][aCol]      = ra1;
  *(uint4*)&Ws[0][aRow0][aCol]      = rw0;
  *(uint4*)&Ws[0][aRow1][aCol]      = rw1;
  *(uint4*)&Ws[0][aRow0 + 64][aCol] = rw2;
  *(uint4*)&Ws[0][aRow1 + 64][aCol] = rw3;
  __syncthreads();

  for (int kt = 0; kt < nk; ++kt) {
    const int cur = kt & 1;
    const int nxt = cur ^ 1;

    // issue next tile's global loads before computing on the current tile
    if (kt + 1 < nk) {
      const int off = (kt + 1) * 64;
      ra0 = *(const uint4*)(Aptr0 + off);
      ra1 = *(const uint4*)(Aptr1 + off);
      rw0 = *(const uint4*)(Wp0 + off);
      rw1 = *(const uint4*)(Wp1 + off);
      rw2 = *(const uint4*)(Wp2 + off);
      rw3 = *(const uint4*)(Wp3 + off);
    }
    if (kt + 2 < nk) {
      const int poff = (kt + 2) * 64;
      __builtin_prefetch(Aptr0 + poff, 0, 1);
      __builtin_prefetch(Aptr1 + poff, 0, 1);
      __builtin_prefetch(Wp0 + poff, 0, 1);
      __builtin_prefetch(Wp1 + poff, 0, 1);
      __builtin_prefetch(Wp2 + poff, 0, 1);
      __builtin_prefetch(Wp3 + poff, 0, 1);
    }

    // --- two 32-deep K slices from the current buffer ---
    const int am0 = wr * 32 + hl;
    const int bn0 = wc * 32 + hl;
#pragma unroll
    for (int s = 0; s < 2; ++s) {
      const int sk = s * 32;
      // A frag (16x32): lane<16 -> row, K 0..7 & 16..23; lane>=16 -> +8.
      const int ak = sk + hi * 8;
      v8h a0lo = *(const v8h*)&As[cur][am0][ak];
      v8h a0hi = *(const v8h*)&As[cur][am0][ak + 16];
      v16h a0 = __builtin_shufflevector(a0lo, a0hi,
          0,1,2,3,4,5,6,7,8,9,10,11,12,13,14,15);
      v8h a1lo = *(const v8h*)&As[cur][am0 + 16][ak];
      v8h a1hi = *(const v8h*)&As[cur][am0 + 16][ak + 16];
      v16h a1 = __builtin_shufflevector(a1lo, a1hi,
          0,1,2,3,4,5,6,7,8,9,10,11,12,13,14,15);

      // B frag (32x16): lane<16 -> col, K 0..7 & 16..23; lane>=16 -> +8.
      const int bk = sk + hi * 8;
      v8h b0a = *(const v8h*)&Ws[cur][bn0][bk];
      v8h b0b = *(const v8h*)&Ws[cur][bn0][bk + 16];
      v16h b0 = __builtin_shufflevector(b0a, b0b,
          0,1,2,3,4,5,6,7,8,9,10,11,12,13,14,15);
      v8h b1a = *(const v8h*)&Ws[cur][bn0 + 16][bk];
      v8h b1b = *(const v8h*)&Ws[cur][bn0 + 16][bk + 16];
      v16h b1 = __builtin_shufflevector(b1a, b1b,
          0,1,2,3,4,5,6,7,8,9,10,11,12,13,14,15);

      acc00 = __builtin_amdgcn_wmma_f32_16x16x32_f16(
          false, a0, false, b0, (short)0, acc00, false, false);
      acc01 = __builtin_amdgcn_wmma_f32_16x16x32_f16(
          false, a0, false, b1, (short)0, acc01, false, false);
      acc10 = __builtin_amdgcn_wmma_f32_16x16x32_f16(
          false, a1, false, b0, (short)0, acc10, false, false);
      acc11 = __builtin_amdgcn_wmma_f32_16x16x32_f16(
          false, a1, false, b1, (short)0, acc11, false, false);
    }

    // stage next tile into the alternate buffer
    if (kt + 1 < nk) {
      *(uint4*)&As[nxt][aRow0][aCol]      = ra0;
      *(uint4*)&As[nxt][aRow1][aCol]      = ra1;
      *(uint4*)&Ws[nxt][aRow0][aCol]      = rw0;
      *(uint4*)&Ws[nxt][aRow1][aCol]      = rw1;
      *(uint4*)&Ws[nxt][aRow0 + 64][aCol] = rw2;
      *(uint4*)&Ws[nxt][aRow1 + 64][aCol] = rw3;
    }
    __syncthreads();
  }

  // C/D layout: VGPR r -> row hi*8+r, col = lane&15 (+16 for the second frag)
  const int outM  = blockM + wr * 32 + hi * 8;
  const int outN0 = blockN + wc * 32 + hl;
  const float bv0 = bias[outN0];
  const float bv1 = bias[outN0 + 16];
#pragma unroll
  for (int r = 0; r < 8; ++r) {
    Cout[(size_t)(outM + r) * Nn + outN0]      = acc00[r] + bv0;
    Cout[(size_t)(outM + r) * Nn + outN0 + 16] = acc01[r] + bv1;
    Cout[(size_t)(outM + 16 + r) * Nn + outN0]      = acc10[r] + bv0;
    Cout[(size_t)(outM + 16 + r) * Nn + outN0 + 16] = acc11[r] + bv1;
  }
}

// ---------------------------------------------------------------------------
// Prep / elementwise kernels
// ---------------------------------------------------------------------------
__global__ __launch_bounds__(256) void k_f32_to_f16(
    const float* __restrict__ in, _Float16* __restrict__ out, int n)
{
  int i = blockIdx.x * 256 + threadIdx.x;
  if (i < n) out[i] = (_Float16)in[i];
}

// src (B,C,N) f32 -> x (B*N, C) f16
__global__ __launch_bounds__(256) void k_prep_x(
    const float* __restrict__ src, _Float16* __restrict__ xh)
{
  int i = blockIdx.x * 256 + threadIdx.x;       // (row, c), c fastest
  int c = i & (CC - 1);
  int row = i >> 8;
  int n = row % NN, b = row / NN;
  xh[i] = (_Float16)src[((size_t)(b * CC + c)) * NN + n];
}

// [qg_b ; kv_b] -> BIAS(1024), and zero-fill ZB(1024)
__global__ __launch_bounds__(256) void k_concat_bias(
    const float* __restrict__ a, const float* __restrict__ b,
    float* __restrict__ out, float* __restrict__ zb)
{
  int i = blockIdx.x * 256 + threadIdx.x;
  if (i < 512) out[i] = a[i];
  else if (i < 1024) out[i] = b[i - 512];
  else zb[i - 1024] = 0.f;
}

__device__ __forceinline__ float safe_pow_relu(float x, float p) {
  return (x > 0.f) ? __expf(p * __logf(x)) : 0.f;
}
__device__ __forceinline__ float softplusf(float x) {
  return (x > 20.f) ? x : log1pf(__expf(x));
}
__device__ __forceinline__ float sigmoidf_(float x) {
  return 1.f / (1.f + __expf(-x));
}

// Consume qgkv GEMM output (BN,1024): produce qp/qn/kp/kn (B,h,N,32), g, v.
__global__ __launch_bounds__(256) void k_attn_pre(
    const float* __restrict__ O, const float* __restrict__ pos,
    const float* __restrict__ scale, const float* __restrict__ power,
    float* __restrict__ QP, float* __restrict__ QN,
    float* __restrict__ KP, float* __restrict__ KN,
    float* __restrict__ G, float* __restrict__ VF)
{
  int i = blockIdx.x * 256 + threadIdx.x;       // (row, c)
  int c = i & (CC - 1);
  int row = i >> 8;
  int n = row % NN, b = row / NN;

  float q = O[(size_t)row * 1024 + c];
  float g = O[(size_t)row * 1024 + 256 + c];
  float k = O[(size_t)row * 1024 + 512 + c] + pos[n * CC + c];
  float v = O[(size_t)row * 1024 + 768 + c];

  float sc = softplusf(scale[c]);
  float pw = 1.f + 4.f * sigmoidf_(power[c]);   // power flat index == c
  float qs = q / sc, ks = k / sc;

  int head = c >> 5, d = c & 31;
  size_t qi = ((size_t)(b * HEADS + head) * NN + n) * HD + d;
  QP[qi] = safe_pow_relu(qs, pw);
  QN[qi] = safe_pow_relu(-qs, pw);
  KP[qi] = safe_pow_relu(ks, pw);
  KN[qi] = safe_pow_relu(-ks, pw);
  G[i]  = g;
  VF[i] = v;
}

// Per (b,h): kmean (64) and kv matrices (2 x 64x16), LDS-chunked over N.
__global__ __launch_bounds__(256) void k_kv_reduce(
    const float* __restrict__ KP, const float* __restrict__ KN,
    const float* __restrict__ VF,
    float* __restrict__ KMEAN, float* __restrict__ KVMAT)
{
  __shared__ float kcS[100][64];
  __shared__ float vS[100][32];
  const int bh = blockIdx.x;             // 0..511
  const int b = bh >> 3, head = bh & 7;
  const int tid = threadIdx.x;
  const int d = tid >> 2;                // 0..63
  const int eb = (tid & 3) * 4;          // 0,4,8,12

  float accS[4] = {0,0,0,0}, accO[4] = {0,0,0,0};
  float ksum = 0.f;

  for (int c0 = 0; c0 < NN; c0 += 100) {
    for (int i = tid; i < 9600; i += 256) {
      if (i < 6400) {
        int n = i >> 6, dd = i & 63;
        size_t idx = ((size_t)bh * NN + c0 + n) * HD + (dd & 31);
        kcS[n][dd] = (dd < 32) ? KP[idx] : KN[idx];
      } else {
        int j = i - 6400;
        int n = j >> 5, e = j & 31;
        int ch = (e < 16) ? (head * 16 + e) : (128 + head * 16 + (e - 16));
        vS[n][e] = VF[((size_t)b * NN + c0 + n) * CC + ch];
      }
    }
    __syncthreads();
    for (int n = 0; n < 100; ++n) {
      float kv = kcS[n][d];
#pragma unroll
      for (int j = 0; j < 4; ++j) {
        accS[j] = fmaf(kv, vS[n][eb + j], accS[j]);
        accO[j] = fmaf(kv, vS[n][16 + eb + j], accO[j]);
      }
    }
    if (tid < 64) {
      float s = 0.f;
      for (int n = 0; n < 100; ++n) s += kcS[n][tid];
      ksum += s;
    }
    __syncthreads();
  }
  const float invN = 1.f / (float)NN;
  size_t base = (size_t)bh * 2048;
#pragma unroll
  for (int j = 0; j < 4; ++j) {
    KVMAT[base + d * 16 + eb + j]        = accS[j] * invN;   // sim
    KVMAT[base + 1024 + d * 16 + eb + j] = accO[j] * invN;   // opp
  }
  if (tid < 64) KMEAN[(size_t)bh * 64 + tid] = ksum * invN;
}

// Per (b,h,n): denominators + q @ kv -> attention output (BN,C).
__global__ __launch_bounds__(256) void k_attn_out(
    const float* __restrict__ QP, const float* __restrict__ QN,
    const float* __restrict__ KMEAN, const float* __restrict__ KVMAT,
    float* __restrict__ AT)
{
  __shared__ float kvS[2048];
  __shared__ float kmS[64];
  const int bh = blockIdx.x / 50;
  const int nchunk = blockIdx.x % 50;
  const int tid = threadIdx.x, lane = tid & 31, wave = tid >> 5;
  const int b = bh >> 3, head = bh & 7;

  for (int i = tid; i < 2112; i += 256) {
    if (i < 2048) kvS[i] = KVMAT[(size_t)bh * 2048 + i];
    else kmS[i - 2048] = KMEAN[(size_t)bh * 64 + (i - 2048)];
  }
  __syncthreads();

  const int n = nchunk * 8 + wave;
  const size_t qb = ((size_t)bh * NN + n) * HD;
  float denom = 1e-6f, o = 0.f;

  if (lane < 16) {                       // x_sim, q_sim = [qp, qn]
    const int j = lane;
#pragma unroll 4
    for (int d = 0; d < 32; ++d) {
      float qp = QP[qb + d], qn = QN[qb + d];
      denom += qp * kmS[d] + qn * kmS[32 + d];
      o += qp * kvS[d * 16 + j] + qn * kvS[(32 + d) * 16 + j];
    }
    AT[((size_t)b * NN + n) * CC + head * 32 + j] = o / denom;
  } else {                               // x_opp, q_opp = [qn, qp]
    const int j = lane - 16;
#pragma unroll 4
    for (int d = 0; d < 32; ++d) {
      float qp = QP[qb + d], qn = QN[qb + d];
      denom += qn * kmS[d] + qp * kmS[32 + d];
      o += qn * kvS[1024 + d * 16 + j] + qp * kvS[1024 + (32 + d) * 16 + j];
    }
    AT[((size_t)b * NN + n) * CC + head * 32 + 16 + j] = o / denom;
  }
}

// Depthwise 5x5 over v reinterpreted per-batch as (8,20,20,32), pad 2.
__global__ __launch_bounds__(256) void k_dwc5(
    const float* __restrict__ VF, const float* __restrict__ dwc_w,
    const float* __restrict__ dwc_b, float* __restrict__ VD)
{
  int i = blockIdx.x * 256 + threadIdx.x;       // over B*102400
  int b = i / (NN * CC);
  int f = i % (NN * CC);
  int r = f / 12800;
  int rem = f % 12800;
  int y = rem / 640;
  int x = (rem / 32) % 20;
  int d = rem & 31;
  const float* vb = VF + (size_t)b * (NN * CC) + r * 12800;
  float acc = dwc_b[d];
#pragma unroll
  for (int ky = 0; ky < 5; ++ky) {
    int yy = y + ky - 2;
    if (yy < 0 || yy >= HH) continue;
#pragma unroll
    for (int kx = 0; kx < 5; ++kx) {
      int xx = x + kx - 2;
      if (xx < 0 || xx >= WW2) continue;
      acc = fmaf(dwc_w[d * 25 + ky * 5 + kx], vb[(yy * 20 + xx) * 32 + d], acc);
    }
  }
  VD[i] = acc;
}

// (attn + vd) * g -> f16 for the projection GEMM
__global__ __launch_bounds__(256) void k_gate(
    const float* __restrict__ AT, const float* __restrict__ VD,
    const float* __restrict__ G, _Float16* __restrict__ AH)
{
  int i = blockIdx.x * 256 + threadIdx.x;
  AH[i] = (_Float16)((AT[i] + VD[i]) * G[i]);
}

// Channel LayerNorm over C: t = src(B,C,N) + add(BN,C); outputs (B,C,N) f32
// and optionally (BN,C) f16.
__global__ __launch_bounds__(256) void k_chan_ln(
    const float* __restrict__ src, const float* __restrict__ add,
    const float* __restrict__ lw, const float* __restrict__ lb,
    float* __restrict__ outCN, _Float16* __restrict__ outRCh)
{
  __shared__ float red[256];
  const int row = blockIdx.x;            // (b,n)
  const int c = threadIdx.x;
  const int b = row / NN, n = row % NN;
  float t = src[(size_t)(b * CC + c) * NN + n] + add[(size_t)row * CC + c];

  red[c] = t; __syncthreads();
  for (int s = 128; s > 0; s >>= 1) { if (c < s) red[c] += red[c + s]; __syncthreads(); }
  float u = red[0] * (1.f / CC);
  __syncthreads();
  float dv = t - u;
  red[c] = dv * dv; __syncthreads();
  for (int s = 128; s > 0; s >>= 1) { if (c < s) red[c] += red[c + s]; __syncthreads(); }
  float var = red[0] * (1.f / CC);
  float sv = lw[c] * dv * rsqrtf(var + 1e-6f) + lb[c];
  outCN[(size_t)(b * CC + c) * NN + n] = sv;
  if (outRCh) outRCh[(size_t)row * CC + c] = (_Float16)sv;
}

// Spatial mean of src per (b,c) -> f16 (B,C) row-major for the SE GEMM.
__global__ __launch_bounds__(256) void k_srcmean(
    const float* __restrict__ src, _Float16* __restrict__ SMH)
{
  int i = blockIdx.x * 256 + threadIdx.x;       // over B*C
  const float* p = src + (size_t)i * NN;
  float s = 0.f;
  for (int n = 0; n < NN; ++n) s += p[n];
  SMH[i] = (_Float16)(s * (1.f / NN));
}

// Depthwise 3x3 on y (BN,1024 channels-last), GELU gate, SE scale -> f16.
__global__ __launch_bounds__(256) void k_sefn_mid(
    const float* __restrict__ Y, const float* __restrict__ dw_w,
    const float* __restrict__ SEPRE, _Float16* __restrict__ YH)
{
  int i = blockIdx.x * 256 + threadIdx.x;       // over BN*512
  int j = i & (HID - 1);
  int row = i >> 9;
  int b = row / NN, n = row % NN;
  int y = n / 20, x = n % 20;
  float a1 = 0.f, a2 = 0.f;
#pragma unroll
  for (int ky = 0; ky < 3; ++ky) {
    int yy = y + ky - 1;
    if (yy < 0 || yy >= HH) continue;
#pragma unroll
    for (int kx = 0; kx < 3; ++kx) {
      int xx = x + kx - 1;
      if (xx < 0 || xx >= WW2) continue;
      size_t base = ((size_t)b * NN + yy * 20 + xx) * 1024;
      a1 = fmaf(dw_w[j * 9 + ky * 3 + kx],         Y[base + j],       a1);
      a2 = fmaf(dw_w[(HID + j) * 9 + ky * 3 + kx], Y[base + HID + j], a2);
    }
  }
  float g1 = 0.5f * a1 * (1.f + erff(a1 * 0.70710678118654752f));
  float se = sigmoidf_(SEPRE[b * HID + j]);
  YH[i] = (_Float16)(g1 * a2 * se);
}

// ---------------------------------------------------------------------------
// Launch
// ---------------------------------------------------------------------------
extern "C" void kernel_launch(void* const* d_in, const int* in_sizes, int n_in,
                              void* d_out, int out_size, void* d_ws, size_t ws_size,
                              hipStream_t stream) {
  const float* src    = (const float*)d_in[0];
  const float* qg_w   = (const float*)d_in[1];
  const float* qg_b   = (const float*)d_in[2];
  const float* kv_w   = (const float*)d_in[3];
  const float* kv_b   = (const float*)d_in[4];
  const float* proj_w = (const float*)d_in[5];
  const float* proj_b = (const float*)d_in[6];
  const float* dwc_w  = (const float*)d_in[7];
  const float* dwc_b  = (const float*)d_in[8];
  const float* power  = (const float*)d_in[9];
  const float* scale  = (const float*)d_in[10];
  const float* pos    = (const float*)d_in[11];
  const float* pin_w  = (const float*)d_in[12];
  const float* dw_w   = (const float*)d_in[13];
  const float* se_w   = (const float*)d_in[14];
  const float* pout_w = (const float*)d_in[15];
  const float* ln1_w  = (const float*)d_in[16];
  const float* ln1_b  = (const float*)d_in[17];
  const float* ln2_w  = (const float*)d_in[18];
  const float* ln2_b  = (const float*)d_in[19];
  float* out = (float*)d_out;

  char* ws = (char*)d_ws;
  _Float16* XH    = (_Float16*)(ws + OFF_XH);
  _Float16* WQGKV = (_Float16*)(ws + OFF_WQGKV);
  _Float16* WPROJ = (_Float16*)(ws + OFF_WPROJ);
  _Float16* WPIN  = (_Float16*)(ws + OFF_WPIN);
  _Float16* WPOUT = (_Float16*)(ws + OFF_WPOUT);
  _Float16* WSE   = (_Float16*)(ws + OFF_WSE);
  float*    BIAS  = (float*)(ws + OFF_BIAS);
  float*    ZB    = (float*)(ws + OFF_ZB);
  float*    GEMMB = (float*)(ws + OFF_GEMMB);
  float*    QP    = (float*)(ws + OFF_QP);
  float*    QN    = (float*)(ws + OFF_QN);
  float*    KP    = (float*)(ws + OFF_KP);
  float*    KN    = (float*)(ws + OFF_KN);
  float*    G     = (float*)(ws + OFF_G);
  float*    VF    = (float*)(ws + OFF_VF);
  float*    KMEAN = (float*)(ws + OFF_KMEAN);
  float*    KVMAT = (float*)(ws + OFF_KVMAT);
  float*    AT    = (float*)(ws + OFF_AT);
  float*    VD    = (float*)(ws + OFF_VD);
  _Float16* AH    = (_Float16*)(ws + OFF_AH);
  float*    PA    = (float*)(ws + OFF_PA);
  float*    S     = (float*)(ws + OFF_S);
  _Float16* SH    = (_Float16*)(ws + OFF_SH);
  _Float16* SMH   = (_Float16*)(ws + OFF_SMH);
  float*    SEPRE = (float*)(ws + OFF_SEPRE);
  _Float16* YH    = (_Float16*)(ws + OFF_YH);

  // --- weight / input conversion ---
  k_prep_x<<<BN, 256, 0, stream>>>(src, XH);
  k_f32_to_f16<<<(512*CC)/256, 256, 0, stream>>>(qg_w,   WQGKV,           512*CC);
  k_f32_to_f16<<<(512*CC)/256, 256, 0, stream>>>(kv_w,   WQGKV + 512*CC,  512*CC);
  k_f32_to_f16<<<(CC*CC)/256,  256, 0, stream>>>(proj_w, WPROJ,           CC*CC);
  k_f32_to_f16<<<(1024*CC)/256,256, 0, stream>>>(pin_w,  WPIN,            1024*CC);
  k_f32_to_f16<<<(CC*HID)/256, 256, 0, stream>>>(pout_w, WPOUT,           CC*HID);
  k_f32_to_f16<<<(HID*CC)/256, 256, 0, stream>>>(se_w,   WSE,             HID*CC);
  k_concat_bias<<<8, 256, 0, stream>>>(qg_b, kv_b, BIAS, ZB);

  // --- fused qg + kv GEMM: (BN,256) x (1024,256)^T ---
  k_wmma_gemm<<<dim3(BN/64, 1024/128), 256, 0, stream>>>(
      XH, WQGKV, BIAS, GEMMB, BN, 1024, CC);

  // --- attention ---
  k_attn_pre<<<BN, 256, 0, stream>>>(GEMMB, pos, scale, power, QP, QN, KP, KN, G, VF);
  k_kv_reduce<<<BB * HEADS, 256, 0, stream>>>(KP, KN, VF, KMEAN, KVMAT);
  k_attn_out<<<BB * HEADS * 50, 256, 0, stream>>>(QP, QN, KMEAN, KVMAT, AT);
  k_dwc5<<<BN, 256, 0, stream>>>(VF, dwc_w, dwc_b, VD);
  k_gate<<<BN, 256, 0, stream>>>(AT, VD, G, AH);

  // --- projection GEMM: (BN,256) x (256,256)^T + proj_b ---
  k_wmma_gemm<<<dim3(BN/64, CC/128), 256, 0, stream>>>(
      AH, WPROJ, proj_b, PA, BN, CC, CC);

  // --- LN1: s = LN(src + a) ---
  k_chan_ln<<<BN, 256, 0, stream>>>(src, PA, ln1_w, ln1_b, S, SH);

  // --- SE path: srcmean -> (64,512) GEMM ---
  k_srcmean<<<(BB*CC)/256, 256, 0, stream>>>(src, SMH);
  k_wmma_gemm<<<dim3(1, HID/128), 256, 0, stream>>>(
      SMH, WSE, ZB, SEPRE, 64, HID, CC);

  // --- SEFN expansion GEMM: (BN,256) x (1024,256)^T (reuses GEMMB) ---
  k_wmma_gemm<<<dim3(BN/64, 1024/128), 256, 0, stream>>>(
      SH, WPIN, ZB, GEMMB, BN, 1024, CC);

  // --- dw3x3 + GELU gate + SE ---
  k_sefn_mid<<<(BN*HID)/256, 256, 0, stream>>>(GEMMB, dw_w, SEPRE, YH);

  // --- SEFN contraction GEMM: (BN,512) x (256,512)^T (reuses PA as f) ---
  k_wmma_gemm<<<dim3(BN/64, CC/128), 256, 0, stream>>>(
      YH, WPOUT, ZB, PA, BN, CC, HID);

  // --- LN2: out = LN(s + f) ---
  k_chan_ln<<<BN, 256, 0, stream>>>(S, PA, ln2_w, ln2_b, out, nullptr);
}
